// AFT_Full_55301998903699
// MI455X (gfx1250) — matrix operand order repaired and checked by reading, compiled, not run on hardware
//
#include <hip/hip_runtime.h>
#include <hip/hip_bf16.h>

typedef __bf16 bf16;
typedef __attribute__((ext_vector_type(16))) __bf16 bf16x16;
typedef __attribute__((ext_vector_type(8)))  __bf16 bf16x8;
typedef __attribute__((ext_vector_type(8)))  float  v8f;

#define TOKENS   73728      // 8*96*96
#define DIMC     512
#define HID      64
#define SEQ      96
#define NBH      768        // 8*96
#define KSZ      589824     // 96*96*64 (per-batch k elements)

// ---- CDNA5 async DMA: global -> LDS (16 bytes), tracked on ASYNCcnt ----
__device__ __forceinline__ void async_copy16(void* lds, const void* gaddr) {
  asm volatile("global_load_async_to_lds_b128 %0, %1, off"
               :: "v"((unsigned)(uintptr_t)lds), "v"(gaddr)
               : "memory");
}
__device__ __forceinline__ void wait_async() {
  asm volatile("s_wait_asynccnt 0x0" ::: "memory");
}

// ---- WMMA fragment helpers (CDNA5 16-bit layouts, wave32) ----
// A 16x32 bf16: lane holds row M=l16, 16 contiguous K (K0=half*16)
__device__ __forceinline__ bf16x16 mk_afrag(const bf16* p) {
  bf16x8 lo = *(const bf16x8*)p;
  bf16x8 hi = *(const bf16x8*)(p + 8);
  bf16x16 r;
#pragma unroll
  for (int i = 0; i < 8; ++i) { r[i] = lo[i]; r[i + 8] = hi[i]; }
  return r;
}
// B 32x16 bf16: lane holds column N=l16, 16 contiguous K (K0=half*16)
__device__ __forceinline__ bf16x16 mk_bfrag(const bf16* p) {
  bf16x8 lo = *(const bf16x8*)p;
  bf16x8 hi = *(const bf16x8*)(p + 8);
  bf16x16 r;
#pragma unroll
  for (int i = 0; i < 8; ++i) { r[i] = lo[i]; r[i + 8] = hi[i]; }
  return r;
}
__device__ __forceinline__ v8f wmma_bf16(bf16x16 a, bf16x16 b, v8f c) {
  return __builtin_amdgcn_wmma_f32_16x16x32_bf16(false, a, false, b, (short)0, c, false, false);
}

// ---------------- prep: pack weights to bf16, exp(bias table) ----------------
__global__ __launch_bounds__(256) void prep_kernel(
    const float* __restrict__ wq, const float* __restrict__ bq,
    const float* __restrict__ wk, const float* __restrict__ bk,
    const float* __restrict__ wv, const float* __restrict__ bv,
    const float* __restrict__ outw_f, const float* __restrict__ wtab,
    bf16* __restrict__ wqkv, float* __restrict__ bqkv,
    bf16* __restrict__ ewb, bf16* __restrict__ outw_b) {
  int t = blockIdx.x * 256 + threadIdx.x;
  if (t < 192 * 512) {
    int r = t >> 9, c = t & 511;
    float w = (r < 64) ? wq[r * 512 + c]
            : (r < 128) ? wk[(r - 64) * 512 + c]
                        : wv[(r - 128) * 512 + c];
    wqkv[t] = (bf16)w;
  }
  if (t < 192)
    bqkv[t] = (t < 64) ? bq[t] : (t < 128) ? bk[t - 64] : bv[t - 128];
  if (t < SEQ * SEQ) ewb[t] = (bf16)__expf(wtab[t]);
  if (t < DIMC * HID) outw_b[t] = (bf16)outw_f[t];
}

// ---------------- QKV projection: (TOKENS x 512) @ (512 x 192) ----------------
// Double-buffered K loop: A staged via regs (needs fp32->bf16 cvt),
// B staged via async global->LDS DMA.
__global__ __launch_bounds__(256) void qkv_kernel(
    const float* __restrict__ x, const bf16* __restrict__ wqkv,
    const float* __restrict__ bqkv,
    float* __restrict__ q, float* __restrict__ k, float* __restrict__ v) {
  __shared__ bf16 As[2][32][72];    // 32 tokens x 64-K chunk, double buffered
  __shared__ bf16 Bs[2][192][72];   // 192 outputs x 64-K chunk, double buffered
  const int t = threadIdx.x;
  const int wave = t >> 5, lane = t & 31, half = lane >> 4, l16 = lane & 15;
  const int tok0 = blockIdx.x * 32;
  const int mi = wave >> 2;            // 0..1
  const int ni0 = (wave & 3) * 3;      // 0,3,6,9 -> 3 N-tiles each
  v8f acc[3] = {};
  const int a_row = t >> 3, a_k = (t & 7) << 3;

  auto stage = [&](int kc, int buf) {
    { // A: fp32 global -> cvt -> LDS
      const float* src = x + (size_t)(tok0 + a_row) * DIMC + kc + a_k;
      float4 f0 = *(const float4*)(src);
      float4 f1 = *(const float4*)(src + 4);
      bf16x8 pk = { (bf16)f0.x, (bf16)f0.y, (bf16)f0.z, (bf16)f0.w,
                    (bf16)f1.x, (bf16)f1.y, (bf16)f1.z, (bf16)f1.w };
      *(bf16x8*)&As[buf][a_row][a_k] = pk;
    }
#pragma unroll
    for (int it = 0; it < 6; ++it) { // B: async DMA, 192x64 bf16
      int idx = t + it * 256;
      int r = idx >> 3, c = (idx & 7) << 3;
      async_copy16(&Bs[buf][r][c], wqkv + (size_t)r * DIMC + kc + c);
    }
  };

  stage(0, 0);
  wait_async();
  __syncthreads();
  for (int kci = 0; kci < 8; ++kci) {
    const int buf = kci & 1;
    if (kci < 7) stage((kci + 1) * 64, buf ^ 1);  // prefetch next chunk
#pragma unroll
    for (int ks = 0; ks < 2; ++ks) {
      const int k0 = ks << 5;
      bf16x16 a = mk_afrag(&As[buf][mi * 16 + l16][k0 + half * 16]);
#pragma unroll
      for (int i = 0; i < 3; ++i) {
        bf16x16 b = mk_bfrag(&Bs[buf][(ni0 + i) * 16 + l16][k0 + half * 16]);
        acc[i] = wmma_bf16(a, b, acc[i]);
      }
    }
    wait_async();
    __syncthreads();
  }
#pragma unroll
  for (int i = 0; i < 3; ++i) {
    int n = (ni0 + i) * 16 + l16;
    float bias = bqkv[n];
    float* dst; int nn;
    if (n < 64)       { dst = q; nn = n; }
    else if (n < 128) { dst = k; nn = n - 64; }
    else              { dst = v; nn = n - 128; }
#pragma unroll
    for (int j = 0; j < 8; ++j) {
      int m = mi * 16 + half * 8 + j;
      dst[(size_t)(tok0 + m) * HID + nn] = acc[i][j] + bias;
    }
  }
}

// ---------------- cross-batch max of k ----------------
__global__ __launch_bounds__(256) void maxk_kernel(const float* __restrict__ k,
                                                   float* __restrict__ maxk) {
  int i = blockIdx.x * 256 + threadIdx.x;
  if (i >= KSZ) return;
  float m = k[i];
#pragma unroll
  for (int b = 1; b < 8; ++b) m = fmaxf(m, k[i + (size_t)b * KSZ]);
  maxk[i] = m;
}

// ---------------- AFT core per (b,h): num/den GEMMs + gating ----------------
__global__ __launch_bounds__(256) void aft_kernel(
    const float* __restrict__ q, const float* __restrict__ k,
    const float* __restrict__ v, const float* __restrict__ maxk,
    const bf16* __restrict__ ewb, bf16* __restrict__ y) {
  __shared__ bf16 Aw[96][104];    // exp(w_bias), row i, K=w
  __shared__ bf16 Bek[64][104];   // exp_k transposed: row d, K=w
  __shared__ bf16 Bekv[64][104];  // exp_k*v transposed
  const int t = threadIdx.x;
  const int wave = t >> 5, lane = t & 31, half = lane >> 4, l16 = lane & 15;
  const int bh = blockIdx.x;           // b*96 + h
  const int h = bh % SEQ;
  const int tok0 = bh * SEQ;
  for (int i = t; i < 96 * 12; i += 256) {  // stage A via async DMA
    int r = i / 12, c = (i % 12) * 8;
    async_copy16(&Aw[r][c], ewb + r * SEQ + c);
  }
  for (int i = t; i < SEQ * HID; i += 256) { // exp_k, exp_k*v (coalesced reads)
    int w = i >> 6, d = i & 63;
    size_t gi = (size_t)(tok0 + w) * HID + d;
    float ek = __expf(k[gi] - maxk[(size_t)(h * SEQ + w) * HID + d]);
    Bek[d][w]  = (bf16)ek;
    Bekv[d][w] = (bf16)(ek * v[gi]);
  }
  wait_async();
  __syncthreads();
  v8f accn[3] = {}, accd[3] = {};
  const int pr0 = wave * 3;            // 24 (mi,nj) pairs, 3 per wave
#pragma unroll
  for (int ks = 0; ks < 3; ++ks) {
    int k0 = ks * 32;
#pragma unroll
    for (int i = 0; i < 3; ++i) {
      int p = pr0 + i, mi = p >> 2, nj = p & 3;
      bf16x16 a  = mk_afrag(&Aw[mi * 16 + l16][k0 + half * 16]);
      bf16x16 bn = mk_bfrag(&Bekv[nj * 16 + l16][k0 + half * 16]);
      bf16x16 bd = mk_bfrag(&Bek [nj * 16 + l16][k0 + half * 16]);
      accn[i] = wmma_bf16(a, bn, accn[i]);
      accd[i] = wmma_bf16(a, bd, accd[i]);
    }
  }
#pragma unroll
  for (int i = 0; i < 3; ++i) {
    int p = pr0 + i, mi = p >> 2, nj = p & 3;
    int d = nj * 16 + l16;
#pragma unroll
    for (int j = 0; j < 8; ++j) {
      int m = mi * 16 + half * 8 + j;
      size_t gi = (size_t)(tok0 + m) * HID + d;
      float sig = 1.0f / (1.0f + __expf(-q[gi]));
      y[gi] = (bf16)(sig * accn[i][j] / accd[i][j]);
    }
  }
}

// ---------------- output projection: (TOKENS x 64) @ (64 x 512) ----------------
__global__ __launch_bounds__(256) void out_kernel(
    const bf16* __restrict__ y, const bf16* __restrict__ outw,
    const float* __restrict__ outb, float* __restrict__ out) {
  __shared__ bf16 Ay[32][72];
  __shared__ bf16 Bw[256][72];
  const int t = threadIdx.x;
  const int wave = t >> 5, lane = t & 31, half = lane >> 4, l16 = lane & 15;
  const int tok0 = blockIdx.x * 32;
  const int nh = blockIdx.y * 256;     // N half (0 or 256)
  { int r = t >> 3, c = (t & 7) << 3;  // stage A: 32x64 bf16 via async DMA
    async_copy16(&Ay[r][c], y + (size_t)(tok0 + r) * HID + c); }
  for (int i = t; i < 256 * 8; i += 256) { // stage B: 256x64 bf16 via async DMA
    int r = i >> 3, c = (i & 7) << 3;
    async_copy16(&Bw[r][c], outw + (size_t)(nh + r) * HID + c);
  }
  wait_async();
  __syncthreads();
  const int mi = wave >> 2, nb = (wave & 3) * 4;  // 4 N-tiles per wave
  v8f acc[4] = {};
#pragma unroll
  for (int ks = 0; ks < 2; ++ks) {
    int k0 = ks * 32;
    bf16x16 a = mk_afrag(&Ay[mi * 16 + l16][k0 + half * 16]);
#pragma unroll
    for (int i = 0; i < 4; ++i) {
      bf16x16 b = mk_bfrag(&Bw[(nb + i) * 16 + l16][k0 + half * 16]);
      acc[i] = wmma_bf16(a, b, acc[i]);
    }
  }
#pragma unroll
  for (int i = 0; i < 4; ++i) {
    int n = nh + (nb + i) * 16 + l16;
    float bias = outb[n];
#pragma unroll
    for (int j = 0; j < 8; ++j) {
      int m = mi * 16 + half * 8 + j;
      out[(size_t)(tok0 + m) * DIMC + n] = acc[i][j] + bias;
    }
  }
}

extern "C" void kernel_launch(void* const* d_in, const int* in_sizes, int n_in,
                              void* d_out, int out_size, void* d_ws, size_t ws_size,
                              hipStream_t stream) {
  const float* x    = (const float*)d_in[0];
  const float* wq_w = (const float*)d_in[1];
  const float* wq_b = (const float*)d_in[2];
  const float* wk_w = (const float*)d_in[3];
  const float* wk_b = (const float*)d_in[4];
  const float* wv_w = (const float*)d_in[5];
  const float* wv_b = (const float*)d_in[6];
  const float* out_w = (const float*)d_in[7];
  const float* out_b = (const float*)d_in[8];
  const float* wtab  = (const float*)d_in[9];

  char* ws = (char*)d_ws;
  size_t o = 0;
  float* q    = (float*)(ws + o); o += (size_t)TOKENS * HID * 4;
  float* k    = (float*)(ws + o); o += (size_t)TOKENS * HID * 4;
  float* v    = (float*)(ws + o); o += (size_t)TOKENS * HID * 4;
  float* maxk = (float*)(ws + o); o += (size_t)KSZ * 4;
  bf16* ybf   = (bf16*)(ws + o);  o += (size_t)TOKENS * HID * 2;
  bf16* wqkv  = (bf16*)(ws + o);  o += (size_t)192 * DIMC * 2;
  float* bqkv = (float*)(ws + o); o += 1024;
  bf16* ewb   = (bf16*)(ws + o);  o += (size_t)SEQ * SEQ * 2;
  bf16* outwb = (bf16*)(ws + o);  o += (size_t)DIMC * HID * 2;

  prep_kernel<<<384, 256, 0, stream>>>(wq_w, wq_b, wk_w, wk_b, wv_w, wv_b,
                                       out_w, wtab, wqkv, bqkv, ewb, outwb);
  qkv_kernel<<<TOKENS / 32, 256, 0, stream>>>(x, wqkv, bqkv, q, k, v);
  maxk_kernel<<<KSZ / 256, 256, 0, stream>>>(k, maxk);
  aft_kernel<<<NBH, 256, 0, stream>>>(q, k, v, maxk, ewb, ybf);
  out_kernel<<<dim3(TOKENS / 32, 2), 256, 0, stream>>>(ybf, outwb, out_b,
                                                       (float*)d_out);
}
